// RewaHierarchicalAttention_90237262889103
// MI455X (gfx1250) — matrix operand order repaired and checked by reading, not compiled, not run on hardware
//
#include <hip/hip_runtime.h>
#include <hip/hip_bf16.h>

typedef __bf16 bf16_t;
typedef __attribute__((ext_vector_type(16))) __bf16 v16bf;
typedef __attribute__((ext_vector_type(8)))  float  v8f;

union FragBf { v16bf v; unsigned int u[8]; };

__device__ __forceinline__ int a_k_of(int vg, int hf) {
  return (vg < 4) ? (vg * 2 + hf * 8) : (16 + (vg - 4) * 2 + hf * 8);
}

// A-matrix 16x32 (MxK) fragment: lane&15 = M row, lane>>4 selects K-half swizzle.
__device__ __forceinline__ void load_a_frag(FragBf& f, const bf16_t* base, int rowStride,
                                            int row, int kBase, int lane) {
  const int r  = row + (lane & 15);
  const int hf = lane >> 4;
  const char* p = (const char*)base + (size_t)r * rowStride * sizeof(bf16_t);
#pragma unroll
  for (int vg = 0; vg < 8; ++vg) {
    const int k = kBase + a_k_of(vg, hf);
    f.u[vg] = *(const unsigned int*)(p + (size_t)k * sizeof(bf16_t));
  }
}

// B-matrix 32x16 (KxN) fragment from n-major LDS (pairs contiguous in K).
__device__ __forceinline__ void load_b_frag(FragBf& f, const bf16_t* base, int rowStride,
                                            int nRow, int kBase, int lane) {
  const int r  = nRow + (lane & 15);
  const int hf = lane >> 4;
  const char* p = (const char*)base + (size_t)r * rowStride * sizeof(bf16_t);
#pragma unroll
  for (int vg = 0; vg < 8; ++vg) {
    const int k = kBase + a_k_of(vg, hf);
    f.u[vg] = *(const unsigned int*)(p + (size_t)k * sizeof(bf16_t));
  }
}

__device__ __forceinline__ v8f wmma_bf16(const FragBf& a, const FragBf& b, v8f c) {
  return __builtin_amdgcn_wmma_f32_16x16x32_bf16(false, a.v, false, b.v, (short)0, c, false, false);
}

__device__ __forceinline__ void store_out(float* p, float v)  { *p = v; }
__device__ __forceinline__ void store_out(bf16_t* p, float v) { *p = (bf16_t)v; }

// LDS byte offset of a shared-memory pointer (flat LDS aperture keeps offset in addr[31:0]).
__device__ __forceinline__ unsigned lds_off(const void* p) {
  return (unsigned)(unsigned long long)p;
}

// 32 bytes global -> LDS, async (two B128 ops, instruction offset applies to both sides).
__device__ __forceinline__ void async_copy32(unsigned ldsDst, const void* gSrc) {
  asm volatile(
      "global_load_async_to_lds_b128 %0, %1, off\n\t"
      "global_load_async_to_lds_b128 %0, %1, off offset:16"
      :: "v"(ldsDst), "v"(gSrc) : "memory");
}

// 128 bytes global -> LDS, async (8x B128).
__device__ __forceinline__ void async_copy128(unsigned ldsDst, const void* gSrc) {
  asm volatile(
      "global_load_async_to_lds_b128 %0, %1, off\n\t"
      "global_load_async_to_lds_b128 %0, %1, off offset:16\n\t"
      "global_load_async_to_lds_b128 %0, %1, off offset:32\n\t"
      "global_load_async_to_lds_b128 %0, %1, off offset:48\n\t"
      "global_load_async_to_lds_b128 %0, %1, off offset:64\n\t"
      "global_load_async_to_lds_b128 %0, %1, off offset:80\n\t"
      "global_load_async_to_lds_b128 %0, %1, off offset:96\n\t"
      "global_load_async_to_lds_b128 %0, %1, off offset:112"
      :: "v"(ldsDst), "v"(gSrc) : "memory");
}

__device__ __forceinline__ void wait_async_le2() {
  asm volatile("s_wait_asynccnt 0x2" ::: "memory");
}
__device__ __forceinline__ void wait_async_zero() {
  asm volatile("s_wait_asynccnt 0x0" ::: "memory");
}

// ---------------------------------------------------------------- converts
__global__ void f32_to_bf16_kernel(const float* __restrict__ src, bf16_t* __restrict__ dst,
                                   size_t n) {
  size_t i = (size_t)blockIdx.x * blockDim.x + threadIdx.x;
  const size_t stride = (size_t)gridDim.x * blockDim.x;
  for (; i < n; i += stride) dst[i] = (bf16_t)src[i];
}

// ---------------------------------------------------------------- GEMM (bf16 in, WMMA, bias)
// C[M,N] = A[M,K] @ B[K,N] + bias[N].  Tile 128x128, BK=32, 8 waves (4M x 2N).
// Double-buffered LDS; A tile staged via async global->LDS, B tile prefetched in regs.
template <typename OutT>
__global__ __launch_bounds__(256) void gemm_bf16_wmma(const bf16_t* __restrict__ A,
                                                      const bf16_t* __restrict__ Bw,
                                                      const float* __restrict__ bias,
                                                      OutT* __restrict__ C,
                                                      int M, int N, int K) {
  __shared__ bf16_t As[2][128][32];   // unpadded: async B128 stores need 16B alignment
  __shared__ bf16_t Bs[2][128][34];   // n-major: Bs[buf][n][k]
  const int tid   = threadIdx.x;
  const int lane  = tid & 31;
  const int wave  = tid >> 5;
  const int waveM = wave >> 1;     // 0..3
  const int waveN = wave & 1;      // 0..1
  const int blockN = blockIdx.x * 128;
  const int blockM = blockIdx.y * 128;

  const v8f zero = {0.f, 0.f, 0.f, 0.f, 0.f, 0.f, 0.f, 0.f};
  v8f acc[2][4];
#pragma unroll
  for (int i = 0; i < 2; ++i)
#pragma unroll
    for (int j = 0; j < 4; ++j) acc[i][j] = zero;

  const int aRow = tid >> 1;
  const int aCol = (tid & 1) * 16;
  const int bK   = tid & 31;
  const int bN0  = (tid >> 5) * 16;

  const int nIter = K / 32;

  // prologue: async A tile 0, B tile 0 into regs
  async_copy32(lds_off(&As[0][aRow][aCol]),
               A + (size_t)(blockM + aRow) * K + 0 + aCol);
  uint4 bq0, bq1;
  {
    const uint4* gp = (const uint4*)(Bw + (size_t)bK * N + blockN + bN0);
    bq0 = gp[0]; bq1 = gp[1];
  }

  for (int it = 0; it < nIter; ++it) {
    const int buf = it & 1;
    const bool hasNext = (it + 1) < nIter;
    uint4 nb0 = {0, 0, 0, 0}, nb1 = {0, 0, 0, 0};
    if (hasNext) {
      const int ktn = (it + 1) * 32;
      async_copy32(lds_off(&As[buf ^ 1][aRow][aCol]),
                   A + (size_t)(blockM + aRow) * K + ktn + aCol);
      const uint4* gp = (const uint4*)(Bw + (size_t)(ktn + bK) * N + blockN + bN0);
      nb0 = gp[0]; nb1 = gp[1];
    }
    // store current B tile transposed (n-major)
    {
      union { uint4 q; bf16_t h[8]; } p0, p1;
      p0.q = bq0; p1.q = bq1;
#pragma unroll
      for (int i = 0; i < 8; ++i) Bs[buf][bN0 + i][bK] = p0.h[i];
#pragma unroll
      for (int i = 0; i < 8; ++i) Bs[buf][bN0 + 8 + i][bK] = p1.h[i];
    }
    // current A tile has landed once <=2 (the prefetched ones) remain outstanding
    if (hasNext) wait_async_le2(); else wait_async_zero();
    __syncthreads();

    FragBf af[2], bfr[4];
#pragma unroll
    for (int ti = 0; ti < 2; ++ti)
      load_a_frag(af[ti], &As[buf][0][0], 32, waveM * 32 + ti * 16, 0, lane);
#pragma unroll
    for (int tj = 0; tj < 4; ++tj)
      load_b_frag(bfr[tj], &Bs[buf][0][0], 34, waveN * 64 + tj * 16, 0, lane);
#pragma unroll
    for (int ti = 0; ti < 2; ++ti)
#pragma unroll
      for (int tj = 0; tj < 4; ++tj) acc[ti][tj] = wmma_bf16(af[ti], bfr[tj], acc[ti][tj]);
    __syncthreads();
    bq0 = nb0; bq1 = nb1;
  }

  const int hf = lane >> 4;
#pragma unroll
  for (int ti = 0; ti < 2; ++ti)
#pragma unroll
    for (int tj = 0; tj < 4; ++tj) {
      const int n = blockN + waveN * 64 + tj * 16 + (lane & 15);
      const float bv = bias ? bias[n] : 0.f;
#pragma unroll
      for (int vg = 0; vg < 8; ++vg) {
        const int m = blockM + waveM * 32 + ti * 16 + vg + 8 * hf;
        store_out(&C[(size_t)m * N + n], acc[ti][tj][vg] + bv);
      }
    }
}

// ---------------------------------------------------------------- RoPE + LSH bucket
__global__ __launch_bounds__(256) void rope_bucket_kernel(const bf16_t* __restrict__ qkv,
                                                          const float* __restrict__ R,
                                                          bf16_t* __restrict__ qr,
                                                          bf16_t* __restrict__ kr,
                                                          bf16_t* __restrict__ vr,
                                                          unsigned int* __restrict__ buckets) {
  const int Tn = 4096, D3 = 3072, Dd = 1024;
  const int lane = threadIdx.x & 31;
  const int warp = blockIdx.x * (blockDim.x >> 5) + (threadIdx.x >> 5);
  const int bh = warp / Tn;
  const int t  = warp % Tn;
  const int b  = bh >> 4;
  const int h  = bh & 15;

  const size_t rowOff = ((size_t)b * Tn + t) * D3 + h * 64;
  const unsigned int* qp = (const unsigned int*)(qkv + rowOff);
  const unsigned int* kp = (const unsigned int*)(qkv + rowOff + Dd);
  const unsigned int* vp = (const unsigned int*)(qkv + rowOff + 2 * Dd);

  union PK { unsigned int u; bf16_t h[2]; };
  PK q2, k2, v2;
  q2.u = qp[lane]; k2.u = kp[lane]; v2.u = vp[lane];

  const int j0 = 2 * lane, j1 = 2 * lane + 1;
  // inv_freq[i] = 10000^(-2i/64) => exp(-i * ln(10000)/32)
  const float f0 = expf(-(float)(j0 & 31) * 0.28782313662425572f);
  const float f1 = expf(-(float)(j1 & 31) * 0.28782313662425572f);
  const float a0 = (float)t * f0, a1 = (float)t * f1;
  const float c0 = cosf(a0), s0 = sinf(a0);
  const float c1 = cosf(a1), s1 = sinf(a1);

  const float q0 = (float)q2.h[0], q1 = (float)q2.h[1];
  const float k0 = (float)k2.h[0], k1 = (float)k2.h[1];
  const float qr0 = q0 * c0 - q1 * s0;
  const float qr1 = q1 * c1 + q0 * s1;
  const float kr0 = k0 * c0 - k1 * s0;
  const float kr1 = k1 * c1 + k0 * s1;

  const size_t hrow = ((size_t)bh * Tn + t) * 64;
  PK o;
  o.h[0] = (bf16_t)qr0; o.h[1] = (bf16_t)qr1;
  ((unsigned int*)(qr + hrow))[lane] = o.u;
  o.h[0] = (bf16_t)kr0; o.h[1] = (bf16_t)kr1;
  ((unsigned int*)(kr + hrow))[lane] = o.u;
  ((unsigned int*)(vr + hrow))[lane] = v2.u;

  // proj[m] = sum_j qrope[j] * R[j,0,m] ; lane m computes one dot via shuffles.
  float acc = 0.f;
#pragma unroll
  for (int src = 0; src < 32; ++src) {
    const float v0 = __shfl(qr0, src, 32);
    const float v1 = __shfl(qr1, src, 32);
    acc += v0 * R[(size_t)(2 * src) * 64 + lane] + v1 * R[(size_t)(2 * src + 1) * 64 + lane];
  }
  // argmax over concat(proj, -proj), first-index tie rules
  float mxv = acc; int mxi = lane;
  float mnv = acc; int mni = lane;
#pragma unroll
  for (int off = 16; off > 0; off >>= 1) {
    const float ov = __shfl_xor(mxv, off, 32);
    const int   oi = __shfl_xor(mxi, off, 32);
    if (ov > mxv || (ov == mxv && oi < mxi)) { mxv = ov; mxi = oi; }
    const float ov2 = __shfl_xor(mnv, off, 32);
    const int   oi2 = __shfl_xor(mni, off, 32);
    if (ov2 < mnv || (ov2 == mnv && oi2 < mni)) { mnv = ov2; mni = oi2; }
  }
  if (lane == 0) {
    const unsigned int bkt = (mxv >= -mnv) ? (unsigned int)mxi : (unsigned int)(32 + mni);
    buckets[(size_t)bh * Tn + t] = bkt;
  }
}

// ---------------------------------------------------------------- stable counting sort per row
__global__ __launch_bounds__(256) void bucket_sort_kernel(const unsigned int* __restrict__ buckets,
                                                          unsigned int* __restrict__ sidx) {
  __shared__ unsigned short cnt[64 * 256];
  __shared__ unsigned int totals[64];
  __shared__ unsigned int base[64];
  const int Tn = 4096;
  const int bh = blockIdx.x;
  const int t  = threadIdx.x;
  const unsigned int* row = buckets + (size_t)bh * Tn;

  for (int i = t; i < 64 * 256; i += 256) cnt[i] = 0;
  __syncthreads();

  const int e0 = t * 16;
  for (int i = 0; i < 16; ++i) cnt[row[e0 + i] * 256 + t]++;
  __syncthreads();

  if (t < 64) {
    unsigned int run = 0;
    for (int tt = 0; tt < 256; ++tt) {
      const unsigned int tmp = cnt[t * 256 + tt];
      cnt[t * 256 + tt] = (unsigned short)run;
      run += tmp;
    }
    totals[t] = run;
  }
  __syncthreads();
  if (t == 0) {
    unsigned int run = 0;
    for (int bkt = 0; bkt < 64; ++bkt) { base[bkt] = run; run += totals[bkt]; }
  }
  __syncthreads();

  unsigned int* out = sidx + (size_t)bh * Tn;
  for (int i = 0; i < 16; ++i) {
    const unsigned int bkt = row[e0 + i];
    const unsigned int dst = base[bkt] + cnt[bkt * 256 + t];
    cnt[bkt * 256 + t]++;
    out[dst] = (unsigned int)(e0 + i);
  }
}

// ---------------------------------------------------------------- chunked LSH attention (WMMA)
// grid (n_chunks=32, BH=64), 128 threads (4 waves). Gathers rows via sorted_idx
// (async global->LDS, per-lane addresses), softmax(QK^T/8)V, scatters to ctx[B*T, D].
__global__ __launch_bounds__(128) void lsh_attention_kernel(const bf16_t* __restrict__ qr,
                                                            const bf16_t* __restrict__ kr,
                                                            const bf16_t* __restrict__ vr,
                                                            const unsigned int* __restrict__ sidx,
                                                            bf16_t* __restrict__ ctx) {
  extern __shared__ char smem[];
  const int Tn = 4096, Dd = 1024;
  bf16_t* qs = (bf16_t*)smem;            // [128][64]
  bf16_t* ks = qs + 128 * 64;            // [128][64]
  bf16_t* vT = ks + 128 * 64;            // [64][128] (n-major V)
  float*  S  = (float*)(vT + 64 * 128);  // [128][128] f32, P bf16 overlaid per-row
  unsigned int* idxL = (unsigned int*)(S + 128 * 128);  // [128]

  const int c  = blockIdx.x;
  const int bh = blockIdx.y;
  const int b  = bh >> 4, h = bh & 15;
  const int tid  = threadIdx.x;
  const int lane = tid & 31;
  const int wave = tid >> 5;
  const v8f zero = {0.f, 0.f, 0.f, 0.f, 0.f, 0.f, 0.f, 0.f};

  idxL[tid] = sidx[(size_t)bh * Tn + c * 128 + tid];
  __syncthreads();

  // Gather one row per thread: q/k rows via async global->LDS, v via VGPR transpose.
  {
    const unsigned int src = idxL[tid];
    const size_t rbase = ((size_t)bh * Tn + src) * 64;
    async_copy128(lds_off(qs + tid * 64), qr + rbase);
    async_copy128(lds_off(ks + tid * 64), kr + rbase);
    union { uint4 q; bf16_t h[8]; } vv;
    const uint4* vg = (const uint4*)(vr + rbase);
#pragma unroll
    for (int i = 0; i < 8; ++i) {
      vv.q = vg[i];
#pragma unroll
      for (int e = 0; e < 8; ++e) vT[(i * 8 + e) * 128 + tid] = vv.h[e];
    }
  }
  wait_async_zero();
  __syncthreads();

  // S = Q K^T (wave handles 32 q-rows)
#pragma unroll
  for (int ti = 0; ti < 2; ++ti)
#pragma unroll
    for (int tj = 0; tj < 8; ++tj) {
      v8f acc = zero;
#pragma unroll
      for (int kk = 0; kk < 64; kk += 32) {
        FragBf a, bfr;
        load_a_frag(a, qs, 64, wave * 32 + ti * 16, kk, lane);
        load_b_frag(bfr, ks, 64, tj * 16, kk, lane);
        acc = wmma_bf16(a, bfr, acc);
      }
      const int hf = lane >> 4;
      const int n  = tj * 16 + (lane & 15);
#pragma unroll
      for (int vg = 0; vg < 8; ++vg)
        S[(size_t)(wave * 32 + ti * 16 + vg + 8 * hf) * 128 + n] = acc[vg];
    }
  __syncthreads();

  // Softmax (thread per row); convert to bf16 P in-place (write-behind is safe).
  {
    float* Sr = S + (size_t)tid * 128;
    const float scale = 0.125f;  // 1/sqrt(64)
    float mx = -3.4e38f;
    for (int i = 0; i < 128; ++i) mx = fmaxf(mx, Sr[i] * scale);
    float sum = 0.f;
    for (int i = 0; i < 128; ++i) { const float e = expf(Sr[i] * scale - mx); Sr[i] = e; sum += e; }
    const float inv = 1.f / sum;
    bf16_t* Pr = (bf16_t*)Sr;  // row stride 256 bf16; first 128 used
    for (int i = 0; i < 128; ++i) { const float e = Sr[i]; Pr[i] = (bf16_t)(e * inv); }
  }
  __syncthreads();

  // O = P V ; scatter straight into unsorted ctx layout.
  const bf16_t* P = (const bf16_t*)S;  // row stride 256
#pragma unroll
  for (int ti = 0; ti < 2; ++ti)
#pragma unroll
    for (int tj = 0; tj < 4; ++tj) {
      v8f acc = zero;
#pragma unroll
      for (int kc = 0; kc < 4; ++kc) {
        FragBf a, bfr;
        load_a_frag(a, P, 256, wave * 32 + ti * 16, kc * 32, lane);
        load_b_frag(bfr, vT, 128, tj * 16, kc * 32, lane);
        acc = wmma_bf16(a, bfr, acc);
      }
      const int hf = lane >> 4;
      const int n  = tj * 16 + (lane & 15);
#pragma unroll
      for (int vg = 0; vg < 8; ++vg) {
        const int m = wave * 32 + ti * 16 + vg + 8 * hf;
        const unsigned int tOrig = idxL[m];
        ctx[((size_t)b * Tn + tOrig) * Dd + h * 64 + n] = (bf16_t)acc[vg];
      }
    }
}

// ---------------------------------------------------------------- launch
extern "C" void kernel_launch(void* const* d_in, const int* in_sizes, int n_in,
                              void* d_out, int out_size, void* d_ws, size_t ws_size,
                              hipStream_t stream) {
  (void)in_sizes; (void)n_in; (void)out_size; (void)ws_size;
  const int Bn = 4, Tn = 4096, Dn = 1024, Hn = 16;
  const int M = Bn * Tn;           // 16384
  const int BH = Bn * Hn;          // 64

  const float* x      = (const float*)d_in[0];
  const float* W_attn = (const float*)d_in[1];
  const float* b_attn = (const float*)d_in[2];
  const float* W_proj = (const float*)d_in[3];
  const float* b_proj = (const float*)d_in[4];
  const float* R      = (const float*)d_in[5];
  float* out = (float*)d_out;

  char* p = (char*)d_ws;
  bf16_t* xb   = (bf16_t*)p;  p += (size_t)M * Dn * 2;
  bf16_t* wab  = (bf16_t*)p;  p += (size_t)Dn * 3 * Dn * 2;
  bf16_t* wpb  = (bf16_t*)p;  p += (size_t)Dn * Dn * 2;
  bf16_t* qkvb = (bf16_t*)p;  p += (size_t)M * 3 * Dn * 2;
  bf16_t* qrw  = (bf16_t*)p;  p += (size_t)BH * Tn * 64 * 2;
  bf16_t* krw  = (bf16_t*)p;  p += (size_t)BH * Tn * 64 * 2;
  bf16_t* vrw  = (bf16_t*)p;  p += (size_t)BH * Tn * 64 * 2;
  unsigned int* buckets = (unsigned int*)p; p += (size_t)BH * Tn * 4;
  unsigned int* sidx    = (unsigned int*)p; p += (size_t)BH * Tn * 4;
  bf16_t* ctx  = (bf16_t*)p;  p += (size_t)M * Dn * 2;

  // 1. converts
  f32_to_bf16_kernel<<<2048, 256, 0, stream>>>(x, xb, (size_t)M * Dn);
  f32_to_bf16_kernel<<<2048, 256, 0, stream>>>(W_attn, wab, (size_t)Dn * 3 * Dn);
  f32_to_bf16_kernel<<<1024, 256, 0, stream>>>(W_proj, wpb, (size_t)Dn * Dn);

  // 2. qkv = x @ W_attn + b_attn  (bf16 out)
  gemm_bf16_wmma<bf16_t><<<dim3(3 * Dn / 128, M / 128), 256, 0, stream>>>(
      xb, wab, b_attn, qkvb, M, 3 * Dn, Dn);

  // 3. RoPE + buckets (one warp per (bh,t); 8 warps/block)
  rope_bucket_kernel<<<(BH * Tn) / 8, 256, 0, stream>>>(qkvb, R, qrw, krw, vrw, buckets);

  // 4. stable counting sort per (b,h) row
  bucket_sort_kernel<<<BH, 256, 0, stream>>>(buckets, sidx);

  // 5. chunked attention with fused gather/scatter
  const size_t attn_lds = (size_t)(128 * 64 * 2) * 2   // qs, ks
                        + (size_t)(64 * 128 * 2)       // vT
                        + (size_t)(128 * 128 * 4)      // S
                        + (size_t)(128 * 4);           // idx
  lsh_attention_kernel<<<dim3(Tn / 128, BH), 128, attn_lds, stream>>>(qrw, krw, vrw, sidx, ctx);

  // 6. out = ctx @ W_proj + b_proj  (f32 out)
  gemm_bf16_wmma<float><<<dim3(Dn / 128, M / 128), 256, 0, stream>>>(
      ctx, wpb, b_proj, out, M, Dn, Dn);
}
